// GNNLinkPredictor_5162550690505
// MI455X (gfx1250) — matrix operand order repaired and checked by reading, not compiled, not run on hardware
//
#include <hip/hip_runtime.h>
#include <hip/hip_bf16.h>

#define NN   50000
#define NE   800000
#define NP   100000
#define INCH 128
#define HIDC 256
#define EMBC 32
#define TOKC 20
#define F0   160   // IN_CH + EMB

typedef __bf16 v16bf __attribute__((ext_vector_type(16)));
typedef __bf16 v8bf  __attribute__((ext_vector_type(8)));
typedef __bf16 v4bf  __attribute__((ext_vector_type(4)));
typedef float  v8f   __attribute__((ext_vector_type(8)));

__device__ __forceinline__ __bf16 f2bf(float f) {
    union { float f; unsigned u; } v; v.f = f;
    unsigned r = v.u + 0x7FFFu + ((v.u >> 16) & 1u);   // RNE
    unsigned short h = (unsigned short)(r >> 16);
    __bf16 o; __builtin_memcpy(&o, &h, 2); return o;
}
__device__ __forceinline__ float bf2f(__bf16 b) {
    unsigned short h; __builtin_memcpy(&h, &b, 2);
    union { unsigned u; float f; } v; v.u = ((unsigned)h) << 16;
    return v.f;
}

// ---------------- utility kernels ----------------
__global__ void zero_f32_v4(float4* __restrict__ p, int n4) {
    int i = blockIdx.x * blockDim.x + threadIdx.x;
    float4 z = {0.f, 0.f, 0.f, 0.f};
    for (; i < n4; i += gridDim.x * blockDim.x) p[i] = z;
}

// copy x [NN,128] f32 into h [NN,160] bf16 (cols 0..127), 4 elements/thread
__global__ void copy_x_kernel(const float4* __restrict__ x4, __bf16* __restrict__ h) {
    int i = blockIdx.x * blockDim.x + threadIdx.x;
    const int n4 = NN * INCH / 4;
    for (; i < n4; i += gridDim.x * blockDim.x) {
        int e = i * 4;
        int node = e / INCH, j = e % INCH;
        float4 v = x4[i];
        v4bf o; o[0] = f2bf(v.x); o[1] = f2bf(v.y); o[2] = f2bf(v.z); o[3] = f2bf(v.w);
        *(v4bf*)(h + (size_t)node * F0 + j) = o;
    }
}

// ---------------- GWAS encoder: one wave per node, lane = out channel ----------------
__global__ __launch_bounds__(256)
void gwas_kernel(const int* __restrict__ tok, const float* __restrict__ sc,
                 const int* __restrict__ cat, const float* __restrict__ temb,
                 const float* __restrict__ cemb, const float* __restrict__ pw,
                 const float* __restrict__ pb, __bf16* __restrict__ h) {
    int lane = threadIdx.x & 31;
    int node = blockIdx.x * 8 + (threadIdx.x >> 5);
    if (node >= NN) return;
    float wcol[41];
#pragma unroll
    for (int k = 0; k < 41; ++k) wcol[k] = pw[k * EMBC + lane];
    float bias = pb[lane];
    float acc = 0.0f, wacc = 0.0f;
    for (int t = 0; t < TOKC; ++t) {
        int  tk = tok[node * TOKC + t];
        float s = sc[node * TOKC + t];
        int  ct = cat[node * TOKC + t];
        float comb = bias;
        const float* te = temb + (size_t)tk * EMBC;
#pragma unroll
        for (int k = 0; k < EMBC; ++k) comb += te[k] * wcol[k];
        const float* ce = cemb + (size_t)ct * 8;
#pragma unroll
        for (int j = 0; j < 8; ++j) comb += ce[j] * wcol[EMBC + j];
        comb += s * wcol[40];
        float w = (tk != 0) ? s : 0.0f;
        acc  += comb * w;
        wacc += w;
    }
    float feat = acc / fmaxf(wacc, 1e-8f);
    h[(size_t)node * F0 + INCH + lane] = f2bf(feat);
}

// ---------------- aggregation ----------------
__global__ void count_kernel(const int* __restrict__ edst, float* __restrict__ cnt) {
    int i = blockIdx.x * blockDim.x + threadIdx.x;
    if (i < NE) atomicAdd(&cnt[edst[i]], 1.0f);
}

// block = edge, thread = 2 features (one u32 read, two f32 atomics)
__global__ void scatter_kernel(const int* __restrict__ esrc, const int* __restrict__ edst,
                               const __bf16* __restrict__ hin, float* __restrict__ agg, int F) {
    int e = blockIdx.x;
    int f = threadIdx.x * 2;
    int s = esrc[e], d = edst[e];
    v4bf dummy; (void)dummy;
    unsigned u = *(const unsigned*)(hin + (size_t)s * F + f);
    __bf16 b0, b1;
    unsigned short h0 = (unsigned short)(u & 0xFFFFu), h1 = (unsigned short)(u >> 16);
    __builtin_memcpy(&b0, &h0, 2); __builtin_memcpy(&b1, &h1, 2);
    float* dst = agg + (size_t)d * F + f;
    atomicAdd(dst,     bf2f(b0));
    atomicAdd(dst + 1, bf2f(b1));
}

// agg/cnt -> bf16 mean, 4 elements/thread (F % 4 == 0)
__global__ void norm_kernel(const float4* __restrict__ agg4, const float* __restrict__ cnt,
                            __bf16* __restrict__ aggb, int F) {
    int i = blockIdx.x * blockDim.x + threadIdx.x;
    int n4 = NN * F / 4;
    for (; i < n4; i += gridDim.x * blockDim.x) {
        int node = (i * 4) / F;
        float inv = 1.0f / fmaxf(cnt[node], 1.0f);
        float4 v = agg4[i];
        v4bf o;
        o[0] = f2bf(v.x * inv); o[1] = f2bf(v.y * inv);
        o[2] = f2bf(v.z * inv); o[3] = f2bf(v.w * inv);
        *(v4bf*)(aggb + (size_t)i * 4) = o;
    }
}

// ---------------- weight pre-pack: f32 [K,256] -> bf16 WMMA B-fragment order ----------------
// fragment index = (ntile * nK + kstep) * 32 + lane; each holds 16 contiguous bf16 (32B).
// lane n = lane&15 -> column ntile*16+n ; lanes0-15 K{0..7,16..23}, lanes16-31 K{8..15,24..31}.
__global__ void pack_w_kernel(const float* __restrict__ W, __bf16* __restrict__ out, int K) {
    int nK = K >> 5;
    int total = 16 * nK * 32;
    int t = blockIdx.x * blockDim.x + threadIdx.x;
    if (t >= total) return;
    int lane = t & 31;
    int ks   = (t >> 5) % nK;
    int nt   = (t >> 5) / nK;
    int kbase = ks * 32 + (lane >> 4) * 8;
    int n     = nt * 16 + (lane & 15);
    __bf16* o = out + (size_t)t * 16;
#pragma unroll
    for (int i = 0; i < 16; ++i)
        o[i] = f2bf(W[(size_t)(kbase + (i < 8 ? i : i + 8)) * HIDC + n]);
}

// ---------------- fused dual-GEMM: out = relu(Aagg@Wl + Ah@Wr + bias), bf16 out ----------------
// grid = 3125 M-tiles; 8 waves/block; each wave computes N-tiles {wave, wave+8}.
__global__ __launch_bounds__(256)
void sage_gemm(const __bf16* __restrict__ Aagg, const __bf16* __restrict__ Ah, int strideA,
               const __bf16* __restrict__ Wlp, const __bf16* __restrict__ Wrp,
               const float* __restrict__ bias, __bf16* __restrict__ out, int K) {
    int lane  = threadIdx.x & 31;
    int wave  = threadIdx.x >> 5;
    int mtile = blockIdx.x;
    int nK = K >> 5;
    int nt0 = wave, nt1 = wave + 8;
    int m  = lane & 15;
    int hf = lane >> 4;
    int row = mtile * 16 + m;

    v8f acc0 = {}, acc1 = {};
    const __bf16* a1b = Aagg + (size_t)row * strideA + hf * 8;
    const __bf16* a2b = Ah   + (size_t)row * strideA + hf * 8;
    const v16bf* bl = (const v16bf*)Wlp;
    const v16bf* br = (const v16bf*)Wrp;

    for (int ks = 0; ks < nK; ++ks) {
        int k0 = ks * 32;
        // A fragments: lanes0-15 K{0..7,16..23}, lanes16-31 K{8..15,24..31}
        v8bf lo1 = *(const v8bf*)(a1b + k0);
        v8bf hi1 = *(const v8bf*)(a1b + k0 + 16);
        v16bf a1 = __builtin_shufflevector(lo1, hi1, 0,1,2,3,4,5,6,7,8,9,10,11,12,13,14,15);
        v8bf lo2 = *(const v8bf*)(a2b + k0);
        v8bf hi2 = *(const v8bf*)(a2b + k0 + 16);
        v16bf a2 = __builtin_shufflevector(lo2, hi2, 0,1,2,3,4,5,6,7,8,9,10,11,12,13,14,15);
        // packed B fragments: single contiguous 32B load each
        v16bf b00 = bl[((size_t)nt0 * nK + ks) * 32 + lane];
        v16bf b01 = br[((size_t)nt0 * nK + ks) * 32 + lane];
        v16bf b10 = bl[((size_t)nt1 * nK + ks) * 32 + lane];
        v16bf b11 = br[((size_t)nt1 * nK + ks) * 32 + lane];
        acc0 = __builtin_amdgcn_wmma_f32_16x16x32_bf16(false, a1, false, b00, (short)0, acc0, false, false);
        acc0 = __builtin_amdgcn_wmma_f32_16x16x32_bf16(false, a2, false, b01, (short)0, acc0, false, false);
        acc1 = __builtin_amdgcn_wmma_f32_16x16x32_bf16(false, a1, false, b10, (short)0, acc1, false, false);
        acc1 = __builtin_amdgcn_wmma_f32_16x16x32_bf16(false, a2, false, b11, (short)0, acc1, false, false);
    }
    // C layout: VGPR r holds M = r + 8*hf, N = ntile*16 + (lane&15)
    int n0 = nt0 * 16 + m, n1 = nt1 * 16 + m;
    float bn0 = bias[n0], bn1 = bias[n1];
#pragma unroll
    for (int r = 0; r < 8; ++r) {
        int orow = mtile * 16 + r + hf * 8;
        out[(size_t)orow * HIDC + n0] = f2bf(fmaxf(acc0[r] + bn0, 0.0f));
        out[(size_t)orow * HIDC + n1] = f2bf(fmaxf(acc1[r] + bn1, 0.0f));
    }
}

// ---------------- pair scoring: one wave per pair, contiguous b128 loads ----------------
__global__ __launch_bounds__(256)
void pair_kernel(const int* __restrict__ src, const int* __restrict__ dst,
                 const __bf16* __restrict__ z, float* __restrict__ out) {
    int lane = threadIdx.x & 31;
    int p = blockIdx.x * 8 + (threadIdx.x >> 5);
    if (p >= NP) return;
    v8bf a = *(const v8bf*)(z + (size_t)src[p] * HIDC + lane * 8);
    v8bf b = *(const v8bf*)(z + (size_t)dst[p] * HIDC + lane * 8);
    float sum = 0.0f;
#pragma unroll
    for (int i = 0; i < 8; ++i) sum += bf2f(a[i]) * bf2f(b[i]);
#pragma unroll
    for (int off = 16; off; off >>= 1) sum += __shfl_xor(sum, off, 32);
    if (lane == 0) out[p] = 1.0f / (1.0f + expf(-sum));
}

// ---------------- launch ----------------
#define ALIGN4K(x) (((x) + 4095) & ~((size_t)4095))

extern "C" void kernel_launch(void* const* d_in, const int* in_sizes, int n_in,
                              void* d_out, int out_size, void* d_ws, size_t ws_size,
                              hipStream_t stream) {
    const float* x     = (const float*)d_in[0];
    const int*   tok   = (const int*)  d_in[1];
    const float* sc    = (const float*)d_in[2];
    const int*   cat   = (const int*)  d_in[3];
    const int*   ei    = (const int*)  d_in[4];
    const int*   psrc  = (const int*)  d_in[5];
    const int*   pdst  = (const int*)  d_in[6];
    const float* temb  = (const float*)d_in[7];
    const float* cemb  = (const float*)d_in[8];
    const float* pw    = (const float*)d_in[9];
    const float* pb    = (const float*)d_in[10];
    const float* Wl0   = (const float*)d_in[11];
    const float* bl0   = (const float*)d_in[12];
    const float* Wr0   = (const float*)d_in[13];
    const float* Wl1   = (const float*)d_in[14];
    const float* bl1   = (const float*)d_in[15];
    const float* Wr1   = (const float*)d_in[16];
    float* out = (float*)d_out;

    const int* esrc = ei;
    const int* edst = ei + NE;

    char* base = (char*)d_ws;
    size_t off = 0;
    float*  cnt  = (float*)(base + off);   off = ALIGN4K(off + (size_t)NN * 4);
    __bf16* h    = (__bf16*)(base + off);  off = ALIGN4K(off + (size_t)NN * F0 * 2);
    float*  agg  = (float*)(base + off);   off = ALIGN4K(off + (size_t)NN * HIDC * 4);
    __bf16* aggb = (__bf16*)(base + off);  off = ALIGN4K(off + (size_t)NN * HIDC * 2);
    __bf16* h1   = (__bf16*)(base + off);  off = ALIGN4K(off + (size_t)NN * HIDC * 2);
    __bf16* zb   = (__bf16*)(base + off);  off = ALIGN4K(off + (size_t)NN * HIDC * 2);
    __bf16* wl0p = (__bf16*)(base + off);  off = ALIGN4K(off + (size_t)F0 * HIDC * 2);
    __bf16* wr0p = (__bf16*)(base + off);  off = ALIGN4K(off + (size_t)F0 * HIDC * 2);
    __bf16* wl1p = (__bf16*)(base + off);  off = ALIGN4K(off + (size_t)HIDC * HIDC * 2);
    __bf16* wr1p = (__bf16*)(base + off);  off = ALIGN4K(off + (size_t)HIDC * HIDC * 2);

    // weights -> bf16, pre-packed into WMMA B-fragment order (conversion fused)
    pack_w_kernel<<<16, 256, 0, stream>>>(Wl0, wl0p, F0);
    pack_w_kernel<<<16, 256, 0, stream>>>(Wr0, wr0p, F0);
    pack_w_kernel<<<16, 256, 0, stream>>>(Wl1, wl1p, HIDC);
    pack_w_kernel<<<16, 256, 0, stream>>>(Wr1, wr1p, HIDC);

    // h = [x | gwas] in bf16, stride 160
    copy_x_kernel<<<2048, 256, 0, stream>>>((const float4*)x, h);
    gwas_kernel<<<(NN + 7) / 8, 256, 0, stream>>>(tok, sc, cat, temb, cemb, pw, pb, h);

    // degree counts (shared by both layers)
    zero_f32_v4<<<64, 256, 0, stream>>>((float4*)cnt, NN / 4);
    count_kernel<<<(NE + 255) / 256, 256, 0, stream>>>(edst, cnt);

    // layer 0 aggregation (F=160)
    zero_f32_v4<<<2048, 256, 0, stream>>>((float4*)agg, NN * F0 / 4);
    scatter_kernel<<<NE, F0 / 2, 0, stream>>>(esrc, edst, h, agg, F0);
    norm_kernel<<<2048, 256, 0, stream>>>((const float4*)agg, cnt, aggb, F0);
    // layer 0 fused GEMM -> h1 (relu, bf16)
    sage_gemm<<<NN / 16, 256, 0, stream>>>(aggb, h, F0, wl0p, wr0p, bl0, h1, F0);

    // layer 1 aggregation (F=256)
    zero_f32_v4<<<4096, 256, 0, stream>>>((float4*)agg, NN * HIDC / 4);
    scatter_kernel<<<NE, HIDC / 2, 0, stream>>>(esrc, edst, h1, agg, HIDC);
    norm_kernel<<<4096, 256, 0, stream>>>((const float4*)agg, cnt, aggb, HIDC);
    // layer 1 fused GEMM -> z (relu, bf16)
    sage_gemm<<<NN / 16, 256, 0, stream>>>(aggb, h1, HIDC, wl1p, wr1p, bl1, zb, HIDC);

    // pair scoring
    pair_kernel<<<(NP + 7) / 8, 256, 0, stream>>>(psrc, pdst, zb, out);
}
